// Partial_ConvNet_with_Swapping_59287728554336
// MI455X (gfx1250) — matrix operand order repaired and checked by reading, not compiled, not run on hardware
//
#include <hip/hip_runtime.h>
#include <hip/hip_bf16.h>

// ---------------------------------------------------------------------------
// CDNA5 (gfx1250) WMMA types
// ---------------------------------------------------------------------------
typedef __attribute__((ext_vector_type(16))) __bf16 v16bf;
typedef __attribute__((ext_vector_type(8)))  __bf16 v8bf;
typedef __attribute__((ext_vector_type(8)))  float  v8f;
typedef __attribute__((ext_vector_type(4)))  int    v4i;

// Async global->LDS staging (CDNA5 ASYNCcnt path), guarded so compile is safe
// on toolchains that lack the builtins.
#if defined(__has_builtin)
#if __has_builtin(__builtin_amdgcn_global_load_async_to_lds_b128) && \
    __has_builtin(__builtin_amdgcn_s_wait_asynccnt)
#define HAVE_ASYNC_LDS 1
#endif
#endif
#ifndef HAVE_ASYNC_LDS
#define HAVE_ASYNC_LDS 0
#endif

#if HAVE_ASYNC_LDS
typedef __attribute__((address_space(1))) v4i* gp4_t;   // global int4*
typedef __attribute__((address_space(3))) v4i* lp4_t;   // LDS int4*
#endif

__device__ __forceinline__ __bf16 f2bf(float f) {
    unsigned u = __builtin_bit_cast(unsigned, f);
    unsigned r = u + 0x7FFFu + ((u >> 16) & 1u);   // round-to-nearest-even
    unsigned short h = (unsigned short)(r >> 16);
    return __builtin_bit_cast(__bf16, h);
}
__device__ __forceinline__ float bf2f(__bf16 b) {
    unsigned short h = __builtin_bit_cast(unsigned short, b);
    unsigned u = ((unsigned)h) << 16;
    return __builtin_bit_cast(float, u);
}

#define BM 64
#define BN 128
#define BK 32

// Small GEMM tile constants for the NLM kernels
#define GM 64
#define GN 32

// ---------------------------------------------------------------------------
// f32 -> bf16 weight conversion (done once per layer before the conv GEMM)
// ---------------------------------------------------------------------------
__global__ void wcvt_kernel(const float* w, __bf16* out, long n) {
    long i = (long)blockIdx.x * 256 + threadIdx.x;
    if (i < n) out[i] = f2bf(w[i]);
}

// ---------------------------------------------------------------------------
// Generic implicit-GEMM partial-conv kernel (bf16 WMMA, f32 accumulate).
// Block tile 64(M) x 128(N), 8 waves, each wave a 32x32 tile => 4 WMMA/K-step.
// Loader folds in: x*mask multiply, channel concat (src1|src2), nearest x2
// upsample of src1 and of the mask. B tile staged via async-to-LDS when
// available (weights pre-converted to bf16; no bounds checks needed since all
// conv K are multiples of 32 and OC multiples of 128).
// ---------------------------------------------------------------------------
struct ConvDesc {
    const float*  src1; const float* src2; const float* mask;
    const __bf16* wgtb;
    float* out;
    int Nb, C1, C2, IH, IW, OH, OW, OC, KH, KW, stride, pad;
    int up1, mask_up;
};

__global__ __launch_bounds__(256) void conv_wmma_kernel(ConvDesc d) {
    __shared__ __align__(32) __bf16 As[BM][BK];
    __shared__ __align__(32) __bf16 Bs[BN][BK];

    const int tid  = threadIdx.x;
    const int lane = tid & 31;
    const int wave = tid >> 5;
    const int wm   = wave & 1;   // M sub-tile (32 rows each)
    const int wn   = wave >> 1;  // N sub-tile (32 cols each, 0..3)

    const int OHW  = d.OH * d.OW;
    const int Mtot = d.Nb * OHW;
    const int Cin  = d.C1 + d.C2;
    const int KHW  = d.KH * d.KW;
    const int K    = Cin * KHW;          // multiple of 32 for all layers
    const int m0   = blockIdx.x * BM;
    const int n0   = blockIdx.y * BN;

    v8f acc[2][2] = {};

    const int a_m = tid >> 2;            // 0..63
    const int a_k = (tid & 3) * 8;       // 0,8,16,24
    const int b_n = tid >> 1;            // 0..127
    const int b_k = (tid & 1) * 16;      // 0,16

    // Decompose this thread's A row once.
    int mrow = m0 + a_m;
    bool mvalid = (mrow < Mtot);
    int nImg = 0, oh = 0, ow = 0;
    if (mvalid) {
        nImg = mrow / OHW;
        int p = mrow - nImg * OHW;
        oh = p / d.OW; ow = p - oh * d.OW;
    }
    const __bf16* wrow = d.wgtb + (long)(n0 + b_n) * K + b_k;

    for (int k0 = 0; k0 < K; k0 += BK) {
        __syncthreads();
        // ---- stage B: pure bf16 copy of the weight tile ----
#if HAVE_ASYNC_LDS
        {
            gp4_t g = (gp4_t)(wrow + k0);
            lp4_t l = (lp4_t)&Bs[b_n][b_k];
            __builtin_amdgcn_global_load_async_to_lds_b128(g, l, 0, 0);
            __builtin_amdgcn_global_load_async_to_lds_b128(g, l, 16, 0);
        }
#else
        {
            const v8bf* g = (const v8bf*)(wrow + k0);
            *(v8bf*)&Bs[b_n][b_k]     = g[0];
            *(v8bf*)&Bs[b_n][b_k + 8] = g[1];
        }
#endif
        // ---- stage A (im2col, masked, concat, optional up2 of src1) ----
        for (int e = 0; e < 8; ++e) {
            int k = k0 + a_k + e;
            float v = 0.f;
            if (mvalid) {
                int ic  = k / KHW;
                int r2  = k - ic * KHW;
                int kh  = r2 / d.KW;
                int kw  = r2 - kh * d.KW;
                int ih  = oh * d.stride - d.pad + kh;
                int iw  = ow * d.stride - d.pad + kw;
                if (ih >= 0 && ih < d.IH && iw >= 0 && iw < d.IW) {
                    float mval = 1.f;
                    if (d.mask) {
                        int mh = ih, mw = iw, MH = d.IH, MW = d.IW;
                        if (d.mask_up) { mh >>= 1; mw >>= 1; MH >>= 1; MW >>= 1; }
                        mval = d.mask[((long)nImg * MH + mh) * MW + mw];
                    }
                    if (mval != 0.f) {
                        float x;
                        if (ic < d.C1) {
                            if (d.up1)
                                x = d.src1[(((long)nImg * d.C1 + ic) * (d.IH >> 1) + (ih >> 1)) * (d.IW >> 1) + (iw >> 1)];
                            else
                                x = d.src1[(((long)nImg * d.C1 + ic) * d.IH + ih) * d.IW + iw];
                        } else {
                            x = d.src2[(((long)nImg * d.C2 + (ic - d.C1)) * d.IH + ih) * d.IW + iw];
                        }
                        v = x * mval;
                    }
                }
            }
            As[a_m][a_k + e] = f2bf(v);
        }
#if HAVE_ASYNC_LDS
        __builtin_amdgcn_s_wait_asynccnt(0);
#endif
        __syncthreads();
        // ---- fragments + 4 WMMA per wave (contiguous k = 16*hi + s in both A and B) ----
        {
            int r  = lane & 15;
            int hi = lane >> 4;
            v16bf a[2], b[2];
#pragma unroll
            for (int mi = 0; mi < 2; ++mi) {
                const __bf16* arow = &As[wm * 32 + mi * 16 + r][0];
                a[mi] = *(const v16bf*)(arow + hi * 16);
            }
#pragma unroll
            for (int ni = 0; ni < 2; ++ni)
                b[ni] = *(const v16bf*)&Bs[wn * 32 + ni * 16 + r][hi * 16];
#pragma unroll
            for (int mi = 0; mi < 2; ++mi)
#pragma unroll
                for (int ni = 0; ni < 2; ++ni)
                    acc[mi][ni] = __builtin_amdgcn_wmma_f32_16x16x32_bf16(
                        false, a[mi], false, b[ni], (short)0, acc[mi][ni], false, false);
        }
    }

    // ---- epilogue: scatter D tiles to NCHW raw buffer ----
#pragma unroll
    for (int mi = 0; mi < 2; ++mi)
#pragma unroll
        for (int ni = 0; ni < 2; ++ni) {
            int col   = n0 + wn * 32 + ni * 16 + (lane & 15);
            int rbase = m0 + wm * 32 + mi * 16 + 8 * (lane >> 4);
            if (col < d.OC) {
                for (int v = 0; v < 8; ++v) {
                    int m = rbase + v;
                    if (m < Mtot) {
                        int n = m / OHW, p = m - n * OHW;
                        d.out[((long)n * d.OC + col) * OHW + p] = acc[mi][ni][v];
                    }
                }
            }
        }
}

// ---------------------------------------------------------------------------
// Mask convolution: mc = conv(mask, ones(k,k)); new_mask = (mc != 0)
// ---------------------------------------------------------------------------
__global__ void mask_conv_kernel(const float* mprev, float* mc, float* nmask,
                                 int Nb, int IH, int IW, int OH, int OW,
                                 int KH, int KW, int stride, int pad, int up) {
    int idx = blockIdx.x * blockDim.x + threadIdx.x;
    int tot = Nb * OH * OW;
    if (idx >= tot) return;
    int n = idx / (OH * OW);
    int p = idx - n * OH * OW;
    int oh = p / OW, ow = p - oh * OW;
    float s = 0.f;
    for (int kh = 0; kh < KH; ++kh)
        for (int kw = 0; kw < KW; ++kw) {
            int ih = oh * stride - pad + kh, iw = ow * stride - pad + kw;
            if (ih >= 0 && ih < IH && iw >= 0 && iw < IW) {
                int mh = ih, mw = iw, MH = IH, MW = IW;
                if (up) { mh >>= 1; mw >>= 1; MH >>= 1; MW >>= 1; }
                s += mprev[((long)n * MH + mh) * MW + mw];
            }
        }
    mc[idx] = s;
    nmask[idx] = (s == 0.f) ? 0.f : 1.f;
}

// ---------------------------------------------------------------------------
// BatchNorm stats over (N,H,W) of the msum-normalized (hole-zeroed) output.
// stats[2c] = mean, stats[2c+1] = rsqrt(var+eps)  (biased var, training mode)
// ---------------------------------------------------------------------------
__global__ __launch_bounds__(256) void bn_stats_kernel(const float* yraw, const float* mc,
                                                       float* stats, int Nb, int C, int OHW,
                                                       float eps) {
    int c = blockIdx.x;
    __shared__ float ssum[256], ssq[256];
    float s = 0.f, q = 0.f;
    long tot = (long)Nb * OHW;
    for (long i = threadIdx.x; i < tot; i += 256) {
        int n = (int)(i / OHW), p = (int)(i - (long)n * OHW);
        float m = mc[(long)n * OHW + p];
        float v = 0.f;
        if (m != 0.f) v = yraw[((long)n * C + c) * OHW + p] / m;
        s += v; q += v * v;
    }
    ssum[threadIdx.x] = s; ssq[threadIdx.x] = q;
    __syncthreads();
    for (int o = 128; o > 0; o >>= 1) {
        if (threadIdx.x < o) { ssum[threadIdx.x] += ssum[threadIdx.x + o]; ssq[threadIdx.x] += ssq[threadIdx.x + o]; }
        __syncthreads();
    }
    if (threadIdx.x == 0) {
        float mean = ssum[0] / (float)tot;
        float var  = ssq[0] / (float)tot - mean * mean;
        stats[2 * c]     = mean;
        stats[2 * c + 1] = rsqrtf(var + eps);
    }
}

// ---------------------------------------------------------------------------
// Fused pconv epilogue: msum normalize (+optional bias), hole zero, BN, act.
// act: 0=none, 1=relu, 2=leaky(0.2)
// ---------------------------------------------------------------------------
__global__ void finish_kernel(const float* yraw, const float* mc, const float* stats,
                              const float* g, const float* bparam, const float* bias,
                              float* out, int Nb, int C, int OHW, int act) {
    long idx = (long)blockIdx.x * 256 + threadIdx.x;
    long tot = (long)Nb * C * OHW;
    if (idx >= tot) return;
    long p = idx % OHW;
    long t = idx / OHW;
    int c = (int)(t % C);
    int n = (int)(t / C);
    float m = mc[(long)n * OHW + p];
    float v;
    if (m == 0.f) v = 0.f;
    else {
        v = yraw[idx];
        if (bias) { float b = bias[c]; v = (v - b) / m + b; }
        else      v = v / m;
    }
    if (stats) v = (v - stats[2 * c]) * stats[2 * c + 1] * g[c] + bparam[c];
    if (act == 1)      v = v > 0.f ? v : 0.f;
    else if (act == 2) v = v >= 0.f ? v : 0.2f * v;
    out[idx] = v;
}

// ---------------------------------------------------------------------------
// NLM: repack h2 (NCHW f32) -> xb[b][pix][c] bf16
// ---------------------------------------------------------------------------
__global__ void xb_kernel(const float* h2, __bf16* xb, int Nb, int C, int Npix) {
    long idx = (long)blockIdx.x * 256 + threadIdx.x;
    long tot = (long)Nb * Npix * C;
    if (idx >= tot) return;
    int c = (int)(idx % C);
    long t = idx / C;
    int i = (int)(t % Npix);
    int b = (int)(t / Npix);
    xb[idx] = f2bf(h2[((long)b * C + c) * Npix + i]);
}

__global__ __launch_bounds__(256) void sq_kernel(const __bf16* xb, float* sq, int C, int Npix) {
    int row = blockIdx.x;                 // b*Npix + i
    __shared__ float red[256];
    const __bf16* x = xb + (long)row * C;
    float s = 0.f;
    for (int c = threadIdx.x; c < C; c += 256) { float v = bf2f(x[c]); s += v * v; }
    red[threadIdx.x] = s;
    __syncthreads();
    for (int o = 128; o > 0; o >>= 1) {
        if (threadIdx.x < o) red[threadIdx.x] += red[threadIdx.x + o];
        __syncthreads();
    }
    if (threadIdx.x == 0) sq[row] = red[0];
}

// ---------------------------------------------------------------------------
// NLM Gram GEMM: D[i][j] = max(sq_i + sq_j - 2 * <x_i, x_j>, 0)   (per batch)
// ---------------------------------------------------------------------------
__global__ __launch_bounds__(256) void gram_kernel(const __bf16* xball, const float* sq,
                                                   float* D, int N, int C) {
    __shared__ __align__(32) __bf16 As[GM][BK];
    __shared__ __align__(32) __bf16 Bs[GN][BK];
    const int b = blockIdx.z;
    const __bf16* x = xball + (long)b * N * C;
    const int tid = threadIdx.x, lane = tid & 31, wave = tid >> 5;
    const int wm = wave & 3, wn = wave >> 2;
    const int m0 = blockIdx.x * GM, n0 = blockIdx.y * GN;
    const int a_m = tid >> 2, a_k = (tid & 3) * 8;
    const int b_n = tid >> 3, b_k = (tid & 7) * 4;
    v8f acc = {};
    for (int k0 = 0; k0 < C; k0 += BK) {
        __syncthreads();
        {
            int m = m0 + a_m;
            for (int e = 0; e < 8; ++e) {
                int k = k0 + a_k + e;
                As[a_m][a_k + e] = (m < N && k < C) ? x[(long)m * C + k] : f2bf(0.f);
            }
        }
        {
            int n = n0 + b_n;
            for (int e = 0; e < 4; ++e) {
                int k = k0 + b_k + e;
                Bs[b_n][b_k + e] = (n < N && k < C) ? x[(long)n * C + k] : f2bf(0.f);
            }
        }
        __syncthreads();
        int r = lane & 15, hi = lane >> 4;
        const __bf16* arow = &As[wm * 16 + r][0];
        v16bf a = *(const v16bf*)(arow + hi * 16);
        v16bf bb = *(const v16bf*)&Bs[wn * 16 + r][hi * 16];
        acc = __builtin_amdgcn_wmma_f32_16x16x32_bf16(false, a, false, bb, (short)0, acc, false, false);
    }
    int j = n0 + wn * 16 + (lane & 15);
    int ibase = m0 + wm * 16 + 8 * (lane >> 4);
    if (j < N) {
        for (int v = 0; v < 8; ++v) {
            int i = ibase + v;
            if (i < N) {
                float dd = sq[(long)b * N + i] + sq[(long)b * N + j] - 2.f * acc[v];
                D[(long)b * N * N + (long)i * N + j] = dd > 0.f ? dd : 0.f;
            }
        }
    }
}

// ---------------------------------------------------------------------------
// NLM weights: dmin over bkg j, w = bkg_j ? exp(-(D-dmin)/h2) : 0 (bf16), rowsum
// ---------------------------------------------------------------------------
__global__ __launch_bounds__(256) void nlm_w_kernel(const float* D, const float* nm,
                                                    __bf16* wb, float* wsum, int N, float h2inv) {
    int b = blockIdx.y, i = blockIdx.x;
    const float* Drow = D + (long)b * N * N + (long)i * N;
    const float* bkg  = nm + (long)b * N;
    __shared__ float red[256];
    float mn = 3.4e38f;
    for (int j = threadIdx.x; j < N; j += 256)
        if (bkg[j] > 0.5f) { float d = Drow[j]; mn = d < mn ? d : mn; }
    red[threadIdx.x] = mn;
    __syncthreads();
    for (int o = 128; o > 0; o >>= 1) {
        if (threadIdx.x < o) { float a = red[threadIdx.x + o]; if (a < red[threadIdx.x]) red[threadIdx.x] = a; }
        __syncthreads();
    }
    float dmin = red[0];
    __syncthreads();
    float s = 0.f;
    __bf16* wrow = wb + (long)b * N * N + (long)i * N;
    for (int j = threadIdx.x; j < N; j += 256) {
        float w = 0.f;
        if (bkg[j] > 0.5f) w = __expf(-(Drow[j] - dmin) * h2inv);
        __bf16 wq = f2bf(w);
        wrow[j] = wq;
        s += bf2f(wq);             // keep sum consistent with quantized weights
    }
    red[threadIdx.x] = s;
    __syncthreads();
    for (int o = 128; o > 0; o >>= 1) {
        if (threadIdx.x < o) red[threadIdx.x] += red[threadIdx.x + o];
        __syncthreads();
    }
    if (threadIdx.x == 0) wsum[(long)b * N + i] = red[0];
}

// ---------------------------------------------------------------------------
// NLM average GEMM: avg[i][c] = (w[i,:] @ x[:,c]) / wsum[i]; keep x_i if bkg.
// Output back in NCHW layout (h2s).
// ---------------------------------------------------------------------------
__global__ __launch_bounds__(256) void nlm_avg_kernel(const __bf16* wball, const __bf16* xball,
                                                      const float* wsum, const float* nm,
                                                      const float* h2, float* out, int N, int C) {
    __shared__ __align__(32) __bf16 As[GM][BK];
    __shared__ __align__(32) __bf16 Bs[GN][BK];
    const int b = blockIdx.z;
    const __bf16* W = wball + (long)b * N * N;
    const __bf16* X = xball + (long)b * N * C;
    const int tid = threadIdx.x, lane = tid & 31, wave = tid >> 5;
    const int wm = wave & 3, wn = wave >> 2;
    const int m0 = blockIdx.x * GM, n0 = blockIdx.y * GN;
    const int a_m = tid >> 2, a_k = (tid & 3) * 8;
    const int b_n = tid >> 3, b_k = (tid & 7) * 4;
    v8f acc = {};
    for (int k0 = 0; k0 < N; k0 += BK) {       // K = N (1296, guarded)
        __syncthreads();
        {
            int m = m0 + a_m;
            for (int e = 0; e < 8; ++e) {
                int k = k0 + a_k + e;
                As[a_m][a_k + e] = (m < N && k < N) ? W[(long)m * N + k] : f2bf(0.f);
            }
        }
        {
            int n = n0 + b_n;                   // column c (C multiple of 32)
            for (int e = 0; e < 4; ++e) {
                int k = k0 + b_k + e;
                Bs[b_n][b_k + e] = (k < N) ? X[(long)k * C + n] : f2bf(0.f);
            }
        }
        __syncthreads();
        int r = lane & 15, hi = lane >> 4;
        const __bf16* arow = &As[wm * 16 + r][0];
        v16bf a = *(const v16bf*)(arow + hi * 16);
        v16bf bb = *(const v16bf*)&Bs[wn * 16 + r][hi * 16];
        acc = __builtin_amdgcn_wmma_f32_16x16x32_bf16(false, a, false, bb, (short)0, acc, false, false);
    }
    int c = n0 + wn * 16 + (lane & 15);
    int ibase = m0 + wm * 16 + 8 * (lane >> 4);
    for (int v = 0; v < 8; ++v) {
        int i = ibase + v;
        if (i < N) {
            float val = acc[v] / wsum[(long)b * N + i];
            if (nm[(long)b * N + i] > 0.5f) val = h2[((long)b * C + c) * N + i];
            out[((long)b * C + c) * N + i] = val;
        }
    }
}

// ---------------------------------------------------------------------------
// Host driver
// ---------------------------------------------------------------------------
extern "C" void kernel_launch(void* const* d_in, const int* in_sizes, int n_in,
                              void* d_out, int out_size, void* d_ws, size_t ws_size,
                              hipStream_t stream) {
    (void)in_sizes; (void)n_in; (void)out_size; (void)ws_size;

    const float* input_x = (const float*)d_in[0];
    const float* mask0   = (const float*)d_in[1];
    const float* enc1_W  = (const float*)d_in[2];
    const float* enc2_W  = (const float*)d_in[3];
    const float* enc3_W  = (const float*)d_in[4];
    const float* enc4_W  = (const float*)d_in[5];
    const float* dec4_W  = (const float*)d_in[6];
    const float* dec3_W  = (const float*)d_in[7];
    const float* dec2_W  = (const float*)d_in[8];
    const float* dec1_W  = (const float*)d_in[9];
    const float* dec1_b  = (const float*)d_in[10];
    const float* enc2_g  = (const float*)d_in[11];
    const float* enc2_bb = (const float*)d_in[12];
    const float* enc3_g  = (const float*)d_in[13];
    const float* enc3_bb = (const float*)d_in[14];
    const float* enc4_g  = (const float*)d_in[15];
    const float* enc4_bb = (const float*)d_in[16];
    const float* dec4_g  = (const float*)d_in[17];
    const float* dec4_bb = (const float*)d_in[18];
    const float* dec3_g  = (const float*)d_in[19];
    const float* dec3_bb = (const float*)d_in[20];
    const float* dec2_g  = (const float*)d_in[21];
    const float* dec2_bb = (const float*)d_in[22];

    const int Nb = 8;
    const int NPIX2 = 36 * 36;       // 1296
    const int C2CH  = 512;

    // ---- workspace carving (256B aligned) ----
    size_t off = 0;
    auto alloc = [&](size_t bytes) -> char* {
        char* p = (char*)d_ws + off;
        off += (bytes + 255) & ~(size_t)255;
        return p;
    };
    float* raw  = (float*)alloc((size_t)Nb * 128 * 144 * 144 * 4);   // max raw (dec1), reused as D
    float* h1   = (float*)alloc((size_t)Nb * 256 * 72 * 72 * 4);
    float* h2   = (float*)alloc((size_t)Nb * 512 * 36 * 36 * 4);
    float* h2s  = (float*)alloc((size_t)Nb * 512 * 36 * 36 * 4);
    float* h3   = (float*)alloc((size_t)Nb * 512 * 18 * 18 * 4);
    float* h4   = (float*)alloc((size_t)Nb * 512 * 9 * 9 * 4);
    float* hd4  = (float*)alloc((size_t)Nb * 512 * 18 * 18 * 4);
    float* hd3  = (float*)alloc((size_t)Nb * 512 * 36 * 36 * 4);
    float* hd2  = (float*)alloc((size_t)Nb * 256 * 72 * 72 * 4);
    float* mc   = (float*)alloc((size_t)Nb * 144 * 144 * 4);
    float* nm1  = (float*)alloc((size_t)Nb * 72 * 72 * 4);
    float* nm2  = (float*)alloc((size_t)Nb * 36 * 36 * 4);
    float* nm3  = (float*)alloc((size_t)Nb * 18 * 18 * 4);
    float* nm4  = (float*)alloc((size_t)Nb * 9 * 9 * 4);
    float* nmd4 = (float*)alloc((size_t)Nb * 18 * 18 * 4);
    float* nmd3 = (float*)alloc((size_t)Nb * 36 * 36 * 4);
    float* nmd2 = (float*)alloc((size_t)Nb * 72 * 72 * 4);
    float* nmd1 = (float*)alloc((size_t)Nb * 144 * 144 * 4);
    float* stats = (float*)alloc(512 * 2 * 4);
    __bf16* xb  = (__bf16*)alloc((size_t)Nb * NPIX2 * C2CH * 2);
    float* sq   = (float*)alloc((size_t)Nb * NPIX2 * 4);
    __bf16* wb  = (__bf16*)alloc((size_t)Nb * NPIX2 * NPIX2 * 2);
    float* wsum = (float*)alloc((size_t)Nb * NPIX2 * 4);
    // bf16 weight copies
    __bf16* Wb_enc1 = (__bf16*)alloc((size_t)256 * 128 * 49 * 2);
    __bf16* Wb_enc2 = (__bf16*)alloc((size_t)512 * 256 * 25 * 2);
    __bf16* Wb_enc3 = (__bf16*)alloc((size_t)512 * 512 * 9 * 2);
    __bf16* Wb_enc4 = (__bf16*)alloc((size_t)512 * 512 * 9 * 2);
    __bf16* Wb_dec4 = (__bf16*)alloc((size_t)512 * 1024 * 9 * 2);
    __bf16* Wb_dec3 = (__bf16*)alloc((size_t)512 * 1024 * 9 * 2);
    __bf16* Wb_dec2 = (__bf16*)alloc((size_t)256 * 768 * 9 * 2);
    __bf16* Wb_dec1 = (__bf16*)alloc((size_t)128 * 384 * 9 * 2);
    float* D    = raw;   // raw buffer is idle between enc2-finish and enc3-conv

    auto wcvt = [&](const float* w, __bf16* dst, long n) {
        wcvt_kernel<<<(unsigned)((n + 255) / 256), 256, 0, stream>>>(w, dst, n);
    };
    wcvt(enc1_W, Wb_enc1, (long)256 * 128 * 49);
    wcvt(enc2_W, Wb_enc2, (long)512 * 256 * 25);
    wcvt(enc3_W, Wb_enc3, (long)512 * 512 * 9);
    wcvt(enc4_W, Wb_enc4, (long)512 * 512 * 9);
    wcvt(dec4_W, Wb_dec4, (long)512 * 1024 * 9);
    wcvt(dec3_W, Wb_dec3, (long)512 * 1024 * 9);
    wcvt(dec2_W, Wb_dec2, (long)256 * 768 * 9);
    wcvt(dec1_W, Wb_dec1, (long)128 * 384 * 9);

    auto conv = [&](const float* s1, int C1, const float* s2, int C2,
                    const float* msk, int mask_up, int up1,
                    const __bf16* w, float* out,
                    int IH, int IW, int OH, int OW, int OC,
                    int KH, int KW, int st, int pd) {
        ConvDesc dsc;
        dsc.src1 = s1; dsc.src2 = s2; dsc.mask = msk; dsc.wgtb = w; dsc.out = out;
        dsc.Nb = Nb; dsc.C1 = C1; dsc.C2 = C2;
        dsc.IH = IH; dsc.IW = IW; dsc.OH = OH; dsc.OW = OW; dsc.OC = OC;
        dsc.KH = KH; dsc.KW = KW; dsc.stride = st; dsc.pad = pd;
        dsc.up1 = up1; dsc.mask_up = mask_up;
        int Mtot = Nb * OH * OW;
        dim3 grid((Mtot + BM - 1) / BM, OC / BN, 1);
        conv_wmma_kernel<<<grid, 256, 0, stream>>>(dsc);
    };
    auto maskc = [&](const float* mprev, float* nmo, int IH, int IW, int OH, int OW,
                     int K, int st, int pd, int up) {
        int tot = Nb * OH * OW;
        mask_conv_kernel<<<(tot + 255) / 256, 256, 0, stream>>>(mprev, mc, nmo,
                                                                Nb, IH, IW, OH, OW, K, K, st, pd, up);
    };
    auto stats_bn = [&](const float* yraw, int C, int OHW) {
        bn_stats_kernel<<<C, 256, 0, stream>>>(yraw, mc, stats, Nb, C, OHW, 1e-5f);
    };
    auto finish = [&](const float* yraw, const float* st, const float* g, const float* bp,
                      const float* bias, float* out, int C, int OHW, int act) {
        long tot = (long)Nb * C * OHW;
        finish_kernel<<<(unsigned)((tot + 255) / 256), 256, 0, stream>>>(yraw, mc, st, g, bp, bias,
                                                                         out, Nb, C, OHW, act);
    };

    // ================= encoder =================
    // enc1: 128 -> 256, 7x7 s2 p3, 144 -> 72, no BN, relu
    conv(input_x, 128, nullptr, 0, mask0, 0, 0, Wb_enc1, raw, 144, 144, 72, 72, 256, 7, 7, 2, 3);
    maskc(mask0, nm1, 144, 144, 72, 72, 7, 2, 3, 0);
    finish(raw, nullptr, nullptr, nullptr, nullptr, h1, 256, 72 * 72, 1);

    // enc2: 256 -> 512, 5x5 s2 p2, 72 -> 36, BN, relu
    conv(h1, 256, nullptr, 0, nm1, 0, 0, Wb_enc2, raw, 72, 72, 36, 36, 512, 5, 5, 2, 2);
    maskc(nm1, nm2, 72, 72, 36, 36, 5, 2, 2, 0);
    stats_bn(raw, 512, 36 * 36);
    finish(raw, stats, enc2_g, enc2_bb, nullptr, h2, 512, 36 * 36, 1);

    // ================= NLM feature swap at layer 2 =================
    {
        long tot = (long)Nb * NPIX2 * C2CH;
        xb_kernel<<<(unsigned)((tot + 255) / 256), 256, 0, stream>>>(h2, xb, Nb, C2CH, NPIX2);
        sq_kernel<<<Nb * NPIX2, 256, 0, stream>>>(xb, sq, C2CH, NPIX2);
        dim3 gg((NPIX2 + GM - 1) / GM, (NPIX2 + GN - 1) / GN, Nb);
        gram_kernel<<<gg, 256, 0, stream>>>(xb, sq, D, NPIX2, C2CH);
        dim3 gw(NPIX2, Nb, 1);
        nlm_w_kernel<<<gw, 256, 0, stream>>>(D, nm2, wb, wsum, NPIX2, 1.0f / 25.0f);
        dim3 ga((NPIX2 + GM - 1) / GM, C2CH / GN, Nb);
        nlm_avg_kernel<<<ga, 256, 0, stream>>>(wb, xb, wsum, nm2, h2, h2s, NPIX2, C2CH);
    }

    // enc3: 512 -> 512, 3x3 s2 p1, 36 -> 18, BN, relu   (input = swapped h2)
    conv(h2s, 512, nullptr, 0, nm2, 0, 0, Wb_enc3, raw, 36, 36, 18, 18, 512, 3, 3, 2, 1);
    maskc(nm2, nm3, 36, 36, 18, 18, 3, 2, 1, 0);
    stats_bn(raw, 512, 18 * 18);
    finish(raw, stats, enc3_g, enc3_bb, nullptr, h3, 512, 18 * 18, 1);

    // enc4: 512 -> 512, 3x3 s2 p1, 18 -> 9, BN, relu
    conv(h3, 512, nullptr, 0, nm3, 0, 0, Wb_enc4, raw, 18, 18, 9, 9, 512, 3, 3, 2, 1);
    maskc(nm3, nm4, 18, 18, 9, 9, 3, 2, 1, 0);
    stats_bn(raw, 512, 9 * 9);
    finish(raw, stats, enc4_g, enc4_bb, nullptr, h4, 512, 9 * 9, 1);

    // ================= decoder =================
    // dec4: concat(up2(h4), h3) 1024 -> 512 @18, BN, leaky
    conv(h4, 512, h3, 512, nm4, 1, 1, Wb_dec4, raw, 18, 18, 18, 18, 512, 3, 3, 1, 1);
    maskc(nm4, nmd4, 18, 18, 18, 18, 3, 1, 1, 1);
    stats_bn(raw, 512, 18 * 18);
    finish(raw, stats, dec4_g, dec4_bb, nullptr, hd4, 512, 18 * 18, 2);

    // dec3: concat(up2(hd4), h2s) 1024 -> 512 @36, BN, leaky
    conv(hd4, 512, h2s, 512, nmd4, 1, 1, Wb_dec3, raw, 36, 36, 36, 36, 512, 3, 3, 1, 1);
    maskc(nmd4, nmd3, 36, 36, 36, 36, 3, 1, 1, 1);
    stats_bn(raw, 512, 36 * 36);
    finish(raw, stats, dec3_g, dec3_bb, nullptr, hd3, 512, 36 * 36, 2);

    // dec2: concat(up2(hd3), h1) 768 -> 256 @72, BN, leaky
    conv(hd3, 512, h1, 256, nmd3, 1, 1, Wb_dec2, raw, 72, 72, 72, 72, 256, 3, 3, 1, 1);
    maskc(nmd3, nmd2, 72, 72, 72, 72, 3, 1, 1, 1);
    stats_bn(raw, 256, 72 * 72);
    finish(raw, stats, dec2_g, dec2_bb, nullptr, hd2, 256, 72 * 72, 2);

    // dec1: concat(up2(hd2), input_x) 384 -> 128 @144, bias, no BN, no act
    conv(hd2, 256, input_x, 128, nmd2, 1, 1, Wb_dec1, raw, 144, 144, 144, 144, 128, 3, 3, 1, 1);
    maskc(nmd2, nmd1, 144, 144, 144, 144, 3, 1, 1, 1);
    finish(raw, nullptr, nullptr, nullptr, dec1_b, (float*)d_out, 128, 144 * 144, 0);
}
